// Attention_45406394253435
// MI455X (gfx1250) — matrix operand order repaired and checked by reading, not compiled, not run on hardware
//
#include <hip/hip_runtime.h>
#include <hip/hip_bf16.h>

#define NPTS 131072
#define BSEG 64
#define CDIM 256
#define DHK  32              // K-tile = DH = 32
#define ATTN_SCALE 0.17677669529663689f   // 1/sqrt(32)

typedef float    v8f   __attribute__((ext_vector_type(8)));
typedef __bf16   v16bf __attribute__((ext_vector_type(16)));
typedef unsigned u32x4 __attribute__((ext_vector_type(4)));
typedef int      i32x4 __attribute__((ext_vector_type(4)));
typedef int      i32x8 __attribute__((ext_vector_type(8)));

union BF16x16 { unsigned u[8]; v16bf v; };

#if defined(__has_builtin)
#if __has_builtin(__builtin_amdgcn_tensor_load_to_lds) && __has_builtin(__builtin_amdgcn_s_wait_tensorcnt)
#define HAVE_TDM 1
#endif
#endif

// order-preserving float<->uint encoding so segment-max can use atomicMax(u32).
// all encodings of real floats are > 0, so init-to-0 acts as -inf.
__device__ __forceinline__ unsigned enc_f32(float x) {
    unsigned u = __float_as_uint(x);
    return (u & 0x80000000u) ? ~u : (u | 0x80000000u);
}
__device__ __forceinline__ float dec_f32(unsigned u) {
    unsigned v = (u & 0x80000000u) ? (u & 0x7fffffffu) : ~u;
    return __uint_as_float(v);
}

#ifdef HAVE_TDM
// TDM load of one 256x32 bf16 weight tile into LDS with row padding 32->34
// elements (pad_interval = 16 DWORDs -> code 3, pad_amount = 1 DWORD -> code 0).
// D# per CDNA5 ISA ch.8: group0 = {count/type, lds_addr, global_addr[56:0]},
// group1 = {flags, tensor dims 256x256, tile dims 32x256, dim0 stride 256}.
// 6-arg builtin form (clang-23 / therock-10.0 headers).
__device__ __forceinline__ void tdm_load_w_tile(const __bf16* gsrc, unsigned lds_off) {
    unsigned long long ga = (unsigned long long)(uintptr_t)gsrc;
    u32x4 g0;
    g0.x = 0x1u;                                           // count=1, user desc
    g0.y = lds_off;                                        // lds_addr (bytes)
    g0.z = (unsigned)(ga & 0xffffffffu);                   // global_addr[31:0]
    g0.w = (unsigned)((ga >> 32) & 0x01ffffffu) | 0x80000000u; // addr[56:32], type=2
    i32x8 g1;
    g1[0] = (1 << 16) | (1 << 20) | (3 << 22);             // data_size=2B, pad_en, 16DW interval, 1DW pad
    g1[1] = (int)(256u << 16);                             // tensor_dim0 = 256 (bits 79:48)
    g1[2] = (int)(256u << 16);                             // tensor_dim1 = 256 (bits 111:80)
    g1[3] = (int)(32u  << 16);                             // tile_dim0  = 32  (bits 127:112)
    g1[4] = 256;                                           // tile_dim1  = 256 (bits 143:128)
    g1[5] = 256;                                           // tensor_dim0_stride = 256
    g1[6] = 0;
    g1[7] = 0;
    i32x4 z4 = {0, 0, 0, 0};
    i32x8 z8 = {0, 0, 0, 0, 0, 0, 0, 0};
    __builtin_amdgcn_tensor_load_to_lds(g0, g1, z4, z4, z8, 0);
}
#endif

// ---------------- prep: weights -> bf16, stats init, tiny k/v projection ----
__global__ void k_prep_weights(const float* __restrict__ Wq,
                               const float* __restrict__ Wo,
                               __bf16* __restrict__ wqb, __bf16* __restrict__ wob) {
    int i = blockIdx.x * 256 + threadIdx.x;
    if (i < CDIM * CDIM) { wqb[i] = (__bf16)Wq[i]; wob[i] = (__bf16)Wo[i]; }
}

__global__ void k_init_stats(unsigned* __restrict__ menc, float* __restrict__ s) {
    int i = blockIdx.x * 256 + threadIdx.x;
    if (i < BSEG * CDIM) { menc[i] = 0u; s[i] = 0.0f; }
}

__global__ void k_kvproj(const float* __restrict__ k, const float* __restrict__ v,
                         const float* __restrict__ Wk, const float* __restrict__ bk,
                         const float* __restrict__ Wv, const float* __restrict__ bv,
                         float* __restrict__ kp, float* __restrict__ vp) {
    const int b = blockIdx.x;      // BSEG blocks
    const int c = threadIdx.x;     // CDIM threads
    float ak = bk[c], av = bv[c];
    const float* kr = k + b * CDIM;
    const float* vr = v + b * CDIM;
    const float* wkr = Wk + c * CDIM;
    const float* wvr = Wv + c * CDIM;
    for (int j = 0; j < CDIM; ++j) {
        ak = fmaf(kr[j], wkr[j], ak);
        av = fmaf(vr[j], wvr[j], av);
    }
    kp[b * CDIM + c] = ak;
    vp[b * CDIM + c] = av;
}

// ---------------- big GEMM #1: qp = q@Wq^T + bq, fused attn + segment max ---
// 8 waves = 4 row-strips x 2 col-halves; each wave: 16 rows x 8 col-tiles
// -> 64 accumulator VGPRs/lane (no spills).
__global__ __launch_bounds__(256, 2)
void k_qp_attn(const float* __restrict__ q, const float* __restrict__ bq,
               const __bf16* __restrict__ wqb, const float* __restrict__ kp,
               const int* __restrict__ batch,
               float* __restrict__ attn, unsigned* __restrict__ menc) {
    __shared__ __bf16 qt[64][DHK + 2];       // A tile, bf16, padded
    __shared__ __bf16 wt[CDIM][DHK + 2];     // B tile (Wq rows), bf16, padded
    __shared__ float  red[4][CDIM];          // per-strip column maxima

    const int tid   = threadIdx.x;
    const int wave  = tid >> 5;
    const int lane  = tid & 31;
    const int half  = lane >> 4;
    const int ln    = lane & 15;
    const int strip = wave >> 1;             // 0..3 row strip
    const int colh  = wave & 1;              // 0..1 column half
    const int row0  = blockIdx.x * 64;

    v8f acc[8];
    for (int t = 0; t < 8; ++t)
        for (int e = 0; e < 8; ++e) acc[t][e] = 0.0f;

    for (int k0 = 0; k0 < CDIM; k0 += DHK) {
#ifdef HAVE_TDM
        if (wave == 0)
            tdm_load_w_tile(wqb + k0, (unsigned)(uintptr_t)&wt[0][0]);
#else
        for (int i = 0; i < 32; ++i) {
            int e = tid + 256 * i, r = e >> 5, kk = e & 31;
            wt[r][kk] = wqb[r * CDIM + k0 + kk];
        }
#endif
        // q tile: 64x32 f32 -> bf16 (8 elements/thread)
        for (int i = 0; i < 8; ++i) {
            int e = tid + 256 * i, r = e >> 5, kk = e & 31;
            qt[r][kk] = (__bf16)q[(size_t)(row0 + r) * CDIM + k0 + kk];
        }
        if (k0 + DHK < CDIM)
            __builtin_prefetch(&q[(size_t)(row0 + (tid >> 2)) * CDIM + k0 + DHK], 0, 0);
#ifdef HAVE_TDM
        if (wave == 0) __builtin_amdgcn_s_wait_tensorcnt(0);
#endif
        __syncthreads();

        // A fragment: 16x32 bf16, ISA lane/K layout (K pairs, half-split)
        BF16x16 af;
        {
            const unsigned* src = (const unsigned*)&qt[strip * 16 + ln][0];
            const int kw = half * 4;
            for (int j = 0; j < 4; ++j) af.u[j]     = src[kw + j];
            for (int j = 0; j < 4; ++j) af.u[4 + j] = src[8 + kw + j];
        }
        // 8 column tiles of B (32x16 bf16): lane = column, K pairs half-split (matches A)
        for (int ct = 0; ct < 8; ++ct) {
            BF16x16 bf_;
            const unsigned* src = (const unsigned*)&wt[colh * 128 + ct * 16 + ln][0];
            for (int j = 0; j < 8; ++j) bf_.u[j] = src[half * 4 + (j & 3) + (j >> 2) * 8];
            acc[ct] = __builtin_amdgcn_wmma_f32_16x16x32_bf16(
                false, af.v, false, bf_.v, (short)0, acc[ct], false, false);
        }
        __syncthreads();
    }

    // epilogue: attn = (qp + bq) * kg * scale ; store ; hierarchical max
    const int gbase = row0 + strip * 16;
    int bidx[8];
    for (int r = 0; r < 8; ++r) bidx[r] = batch[gbase + half * 8 + r];
    const bool uni = (batch[row0] == batch[row0 + 63]);

    for (int ct = 0; ct < 8; ++ct) {
        const int col = colh * 128 + ct * 16 + ln;
        const float bqv = bq[col];
        float lmax = -3.402823466e+38f;
        for (int r = 0; r < 8; ++r) {
            const int g = gbase + half * 8 + r;
            const int b = bidx[r];
            float a = (acc[ct][r] + bqv) * kp[b * CDIM + col] * ATTN_SCALE;
            attn[(size_t)g * CDIM + col] = a;
            if (uni) lmax = fmaxf(lmax, a);
            else     atomicMax(&menc[b * CDIM + col], enc_f32(a));
        }
        if (uni) {
            lmax = fmaxf(lmax, __shfl_xor(lmax, 16, 32));  // combine row halves
            red[strip][col] = lmax;
        }
    }
    if (uni) {   // block-uniform condition: barrier is safe
        __syncthreads();
        float m = red[0][tid];
        for (int w = 1; w < 4; ++w) m = fmaxf(m, red[w][tid]);
        const int b0 = batch[row0];
        atomicMax(&menc[b0 * CDIM + tid], enc_f32(m));
    }
}

// ---------------- pass: e = exp(attn - m), segment sum ---------------------
__global__ void k_expsum(const int* __restrict__ batch,
                         const unsigned* __restrict__ menc,
                         float* __restrict__ attn, float* __restrict__ s) {
    const int col  = threadIdx.x;
    const int row0 = blockIdx.x * 128;
    const bool uni = (batch[row0] == batch[row0 + 127]);
    if (uni) {
        const int b = batch[row0];
        const float mv = dec_f32(menc[b * CDIM + col]);
        float acc = 0.0f;
        for (int r = 0; r < 128; ++r) {
            size_t idx = (size_t)(row0 + r) * CDIM + col;
            float ev = __expf(attn[idx] - mv);
            attn[idx] = ev;
            acc += ev;
        }
        atomicAdd(&s[b * CDIM + col], acc);
    } else {
        for (int r = 0; r < 128; ++r) {
            const int b = batch[row0 + r];
            size_t idx = (size_t)(row0 + r) * CDIM + col;
            float ev = __expf(attn[idx] - dec_f32(menc[b * CDIM + col]));
            attn[idx] = ev;
            atomicAdd(&s[b * CDIM + col], ev);
        }
    }
}

// fold vp/s into a single scale table (s is overwritten)
__global__ void k_finalize(const float* __restrict__ vp, float* __restrict__ s) {
    int i = blockIdx.x * 256 + threadIdx.x;
    if (i < BSEG * CDIM) s[i] = vp[i] / s[i];
}

// ---------------- big GEMM #2: out = (e * vp/s) @ Wo^T + bo ----------------
__global__ __launch_bounds__(256, 2)
void k_out_gemm(const float* __restrict__ e, const float* __restrict__ scale,
                const int* __restrict__ batch, const __bf16* __restrict__ wob,
                const float* __restrict__ bo, float* __restrict__ out) {
    __shared__ __bf16 qt[64][DHK + 2];
    __shared__ __bf16 wt[CDIM][DHK + 2];

    const int tid   = threadIdx.x;
    const int wave  = tid >> 5;
    const int lane  = tid & 31;
    const int half  = lane >> 4;
    const int ln    = lane & 15;
    const int strip = wave >> 1;
    const int colh  = wave & 1;
    const int row0  = blockIdx.x * 64;

    v8f acc[8];
    for (int t = 0; t < 8; ++t)
        for (int ee = 0; ee < 8; ++ee) acc[t][ee] = 0.0f;

    for (int k0 = 0; k0 < CDIM; k0 += DHK) {
#ifdef HAVE_TDM
        if (wave == 0)
            tdm_load_w_tile(wob + k0, (unsigned)(uintptr_t)&wt[0][0]);
#else
        for (int i = 0; i < 32; ++i) {
            int el = tid + 256 * i, r = el >> 5, kk = el & 31;
            wt[r][kk] = wob[r * CDIM + k0 + kk];
        }
#endif
        for (int i = 0; i < 8; ++i) {
            int el = tid + 256 * i, r = el >> 5, kk = el & 31;
            const int g = row0 + r;
            const int b = batch[g];
            float y = e[(size_t)g * CDIM + k0 + kk] * scale[b * CDIM + k0 + kk];
            qt[r][kk] = (__bf16)y;
        }
        if (k0 + DHK < CDIM)
            __builtin_prefetch(&e[(size_t)(row0 + (tid >> 2)) * CDIM + k0 + DHK], 0, 0);
#ifdef HAVE_TDM
        if (wave == 0) __builtin_amdgcn_s_wait_tensorcnt(0);
#endif
        __syncthreads();

        BF16x16 af;
        {
            const unsigned* src = (const unsigned*)&qt[strip * 16 + ln][0];
            const int kw = half * 4;
            for (int j = 0; j < 4; ++j) af.u[j]     = src[kw + j];
            for (int j = 0; j < 4; ++j) af.u[4 + j] = src[8 + kw + j];
        }
        for (int ct = 0; ct < 8; ++ct) {
            BF16x16 bf_;
            const unsigned* src = (const unsigned*)&wt[colh * 128 + ct * 16 + ln][0];
            for (int j = 0; j < 8; ++j) bf_.u[j] = src[half * 4 + (j & 3) + (j >> 2) * 8];
            acc[ct] = __builtin_amdgcn_wmma_f32_16x16x32_bf16(
                false, af.v, false, bf_.v, (short)0, acc[ct], false, false);
        }
        __syncthreads();
    }

    const int gbase = row0 + strip * 16;
    for (int ct = 0; ct < 8; ++ct) {
        const int col = colh * 128 + ct * 16 + ln;
        const float bov = bo[col];
        for (int r = 0; r < 8; ++r) {
            const int g = gbase + half * 8 + r;
            out[(size_t)g * CDIM + col] = acc[ct][r] + bov;
        }
    }
}

// ---------------- host launcher --------------------------------------------
extern "C" void kernel_launch(void* const* d_in, const int* in_sizes, int n_in,
                              void* d_out, int out_size, void* d_ws, size_t ws_size,
                              hipStream_t stream) {
    const float* q  = (const float*)d_in[0];
    const float* k  = (const float*)d_in[1];
    const float* v  = (const float*)d_in[2];
    const int* batch = (const int*)d_in[3];
    const float* Wq = (const float*)d_in[4];
    const float* bq = (const float*)d_in[5];
    const float* Wk = (const float*)d_in[6];
    const float* bk = (const float*)d_in[7];
    const float* Wv = (const float*)d_in[8];
    const float* bv = (const float*)d_in[9];
    const float* Wo = (const float*)d_in[10];
    const float* bo = (const float*)d_in[11];

    char* w = (char*)d_ws;
    float*    attn = (float*)w;                                  // N*C f32 (attn, then e)
    float*    kp   = (float*)(w + (size_t)NPTS * CDIM * 4);      // B*C
    float*    vp   = kp + BSEG * CDIM;                           // B*C
    unsigned* menc = (unsigned*)(vp + BSEG * CDIM);              // B*C
    float*    s    = (float*)(menc + BSEG * CDIM);               // B*C (becomes vp/s)
    __bf16*   wqb  = (__bf16*)(s + BSEG * CDIM);                 // C*C bf16
    __bf16*   wob  = wqb + CDIM * CDIM;                          // C*C bf16

    k_prep_weights<<<(CDIM * CDIM + 255) / 256, 256, 0, stream>>>(Wq, Wo, wqb, wob);
    k_init_stats<<<(BSEG * CDIM + 255) / 256, 256, 0, stream>>>(menc, s);
    k_kvproj<<<BSEG, CDIM, 0, stream>>>(k, v, Wk, bk, Wv, bv, kp, vp);
    k_qp_attn<<<NPTS / 64, 256, 0, stream>>>(q, bq, wqb, kp, batch, attn, menc);
    k_expsum<<<NPTS / 128, 256, 0, stream>>>(batch, menc, attn, s);
    k_finalize<<<(BSEG * CDIM + 255) / 256, 256, 0, stream>>>(vp, s);
    k_out_gemm<<<NPTS / 64, 256, 0, stream>>>(attn, s, batch, wob, bo, (float*)d_out);
}
